// ParameterSharingTransformerLayer_13932873908439
// MI455X (gfx1250) — compile-verified, matrix operand order checked by reading, not run on hardware
//
#include <hip/hip_runtime.h>
#include <hip/hip_bf16.h>
#include <math.h>

#define Bc 8
#define Sc 1024
#define Hc 1024
#define NHc 16
#define HDc 64
#define MBANKc 128
#define TOPKc 16
#define FFc 4096
#define EPSc 1e-5f

typedef __attribute__((ext_vector_type(16))) __bf16 v16bf;
typedef __attribute__((ext_vector_type(8)))  float  v8f;

union Frag { v16bf v; unsigned int u[8]; };

__device__ __forceinline__ unsigned short f2bf(float x) {
    unsigned int u = __float_as_uint(x);
    u = u + 0x7FFFu + ((u >> 16) & 1u);          // round to nearest even
    return (unsigned short)(u >> 16);
}
__device__ __forceinline__ unsigned int pack2bf(float a, float b) {
    return (unsigned int)f2bf(a) | ((unsigned int)f2bf(b) << 16);
}
// 16-bit A/B-operand K pattern: VGPR j, half = lane>>4 -> K base
__device__ __forceinline__ int kpat(int j, int hv) {
    return ((j >> 2) << 4) + (hv << 3) + ((j & 3) << 1);
}

// ---------------------------------------------------------------------------
// Generic GEMM: C[M,N] = act(A[M,K] @ W[N,K]^T + bias) + resid
// 128 threads (4 waves), 64x64 tile, K steps of 32, bf16 WMMA, fp32 accum.
// act: 0 none, 1 relu, 2 gelu(exact erf)
// ---------------------------------------------------------------------------
#define GT_M 64
#define GT_N 64
#define GT_K 32
#define LPITCH 40   // bf16 elements per LDS row (pad vs 32)

__global__ __launch_bounds__(128)
void gemm_bf16_wmma(const float* __restrict__ A, int lda,
                    const float* __restrict__ W, int ldw,
                    const float* __restrict__ bias,
                    const float* __restrict__ resid, int ldr,
                    float* __restrict__ C, int ldc,
                    int K, int act)
{
    __shared__ unsigned short sA[GT_M * LPITCH];
    __shared__ unsigned short sB[GT_N * LPITCH];

    const int tid  = threadIdx.x;
    const int wave = tid >> 5;
    const int lane = tid & 31;
    const int hv   = lane >> 4;
    const int l16  = lane & 15;
    const int bm   = blockIdx.y * GT_M;
    const int bn   = blockIdx.x * GT_N;
    const int wm   = (wave >> 1) * 32;
    const int wn   = (wave & 1) * 32;

    v8f acc[2][2];
    for (int i = 0; i < 2; ++i)
        for (int j = 0; j < 2; ++j)
            for (int e = 0; e < 8; ++e) acc[i][j][e] = 0.0f;

    for (int k0 = 0; k0 < K; k0 += GT_K) {
        __syncthreads();
        // stage A tile (64x32) and W tile (64x32) as bf16 in LDS
        for (int i = tid; i < GT_M * (GT_K / 2); i += 128) {
            int r  = i >> 4;            // 16 pairs per row
            int c2 = (i & 15) << 1;
            const float* ap = A + (size_t)(bm + r) * lda + k0 + c2;
            *(unsigned int*)&sA[r * LPITCH + c2] = pack2bf(ap[0], ap[1]);
            const float* wp = W + (size_t)(bn + r) * ldw + k0 + c2;
            *(unsigned int*)&sB[r * LPITCH + c2] = pack2bf(wp[0], wp[1]);
        }
        __syncthreads();

        Frag af[2], bf[2];
        for (int i = 0; i < 2; ++i) {
            int m = wm + i * 16 + l16;
            for (int j = 0; j < 8; ++j)
                af[i].u[j] = *(const unsigned int*)&sA[m * LPITCH + kpat(j, hv)];
        }
        for (int j2 = 0; j2 < 2; ++j2) {
            int n = wn + j2 * 16 + l16;
            for (int j = 0; j < 8; ++j)
                bf[j2].u[j] = *(const unsigned int*)&sB[n * LPITCH + kpat(j, hv)];
        }
        for (int i = 0; i < 2; ++i)
            for (int j2 = 0; j2 < 2; ++j2)
                acc[i][j2] = __builtin_amdgcn_wmma_f32_16x16x32_bf16(
                    false, af[i].v, false, bf[j2].v, (short)0, acc[i][j2], false, false);
    }

    for (int i = 0; i < 2; ++i)
        for (int j2 = 0; j2 < 2; ++j2) {
            int col = bn + wn + j2 * 16 + l16;
            for (int e = 0; e < 8; ++e) {
                int row = bm + wm + i * 16 + e + hv * 8;
                float v = acc[i][j2][e];
                if (bias) v += bias[col];
                if (act == 1) v = fmaxf(v, 0.0f);
                else if (act == 2) v = 0.5f * v * (1.0f + erff(v * 0.70710678118654752f));
                if (resid) v += resid[(size_t)row * ldr + col];
                C[(size_t)row * ldc + col] = v;
            }
        }
}

// ---------------------------------------------------------------------------
// Flash attention: one wave per 16-query tile, non-causal, scale 1/8.
// qkv layout: row stride ld, q/k/v at column offsets qoff/koff/voff (+head*64)
// ---------------------------------------------------------------------------
__global__ __launch_bounds__(32)
void flash_attn_wmma(const float* __restrict__ qkv, int ld,
                     int qoff, int koff, int voff,
                     float* __restrict__ out, int ldo)
{
    __shared__ unsigned short sP[16 * 32];

    const int qt   = blockIdx.x;
    const int head = blockIdx.y;
    const int b    = blockIdx.z;
    const int lane = threadIdx.x;
    const int hv   = lane >> 4;
    const int l16  = lane & 15;

    const float* qp = qkv + (size_t)b * Sc * ld + qoff + head * HDc;
    const float* kp = qkv + (size_t)b * Sc * ld + koff + head * HDc;
    const float* vp = qkv + (size_t)b * Sc * ld + voff + head * HDc;

    // Q fragment (16x64 -> two k=32 chunks), resident in registers
    Frag qf[2];
    {
        int m = qt * 16 + l16;
        for (int c = 0; c < 2; ++c)
            for (int j = 0; j < 8; ++j) {
                int k = c * 32 + kpat(j, hv);
                const float* p = qp + (size_t)m * ld + k;
                qf[c].u[j] = pack2bf(p[0], p[1]);
            }
    }

    float rmax[8], rsum[8];
    v8f o[4];
    for (int e = 0; e < 8; ++e) { rmax[e] = -3.4e38f; rsum[e] = 0.0f; }
    for (int nt = 0; nt < 4; ++nt)
        for (int e = 0; e < 8; ++e) o[nt][e] = 0.0f;

    for (int kb = 0; kb < Sc / 32; ++kb) {
        // scores: two 16x16 tiles (32 keys)
        v8f sc[2];
        for (int t = 0; t < 2; ++t) {
            for (int e = 0; e < 8; ++e) sc[t][e] = 0.0f;
            Frag kf[2];
            int kr = kb * 32 + t * 16 + l16;   // B operand: lane = key (col of K^T)
            for (int c = 0; c < 2; ++c)
                for (int j = 0; j < 8; ++j) {
                    int k = c * 32 + kpat(j, hv);
                    const float* p = kp + (size_t)kr * ld + k;
                    kf[c].u[j] = pack2bf(p[0], p[1]);
                }
            sc[t] = __builtin_amdgcn_wmma_f32_16x16x32_bf16(
                false, qf[0].v, false, kf[0].v, (short)0, sc[t], false, false);
            sc[t] = __builtin_amdgcn_wmma_f32_16x16x32_bf16(
                false, qf[1].v, false, kf[1].v, (short)0, sc[t], false, false);
        }

        // online softmax (rows owned by 16-lane halves match C-layout)
        float alpha[8];
        for (int e = 0; e < 8; ++e) {
            float s0 = sc[0][e] * 0.125f;
            float s1 = sc[1][e] * 0.125f;
            float mx = fmaxf(s0, s1);
            for (int off = 8; off >= 1; off >>= 1)
                mx = fmaxf(mx, __shfl_xor(mx, off, 16));
            float mnew = fmaxf(rmax[e], mx);
            alpha[e] = __expf(rmax[e] - mnew);
            rmax[e] = mnew;
            float p0 = __expf(s0 - mnew);
            float p1 = __expf(s1 - mnew);
            sc[0][e] = p0; sc[1][e] = p1;
            float sm = p0 + p1;
            for (int off = 8; off >= 1; off >>= 1)
                sm += __shfl_xor(sm, off, 16);
            rsum[e] = rsum[e] * alpha[e] + sm;
        }
        for (int nt = 0; nt < 4; ++nt)
            for (int e = 0; e < 8; ++e) o[nt][e] *= alpha[e];

        // P (16x32) -> LDS in bf16, reload as A-operand fragment
        for (int t = 0; t < 2; ++t)
            for (int e = 0; e < 8; ++e) {
                int row = e + hv * 8;
                int col = l16 + t * 16;
                sP[row * 32 + col] = f2bf(sc[t][e]);
            }
        __syncthreads();
        Frag pf;
        for (int j = 0; j < 8; ++j)
            pf.u[j] = *(const unsigned int*)&sP[l16 * 32 + kpat(j, hv)];

        // O += P (16x32) @ V (32x64)
        for (int nt = 0; nt < 4; ++nt) {
            Frag vf;
            int col = nt * 16 + l16;           // B operand: lane = output column
            for (int j = 0; j < 8; ++j) {
                int k = kpat(j, hv);           // rows of V within this 32-block
                float a = vp[(size_t)(kb * 32 + k) * ld + col];
                float bb = vp[(size_t)(kb * 32 + k + 1) * ld + col];
                vf.u[j] = pack2bf(a, bb);
            }
            o[nt] = __builtin_amdgcn_wmma_f32_16x16x32_bf16(
                false, pf.v, false, vf.v, (short)0, o[nt], false, false);
        }
        __syncthreads();
    }

    for (int nt = 0; nt < 4; ++nt)
        for (int e = 0; e < 8; ++e) {
            int row = b * Sc + qt * 16 + e + hv * 8;
            int col = head * HDc + nt * 16 + l16;
            out[(size_t)row * ldo + col] = o[nt][e] / rsum[e];
        }
}

// ---------------------------------------------------------------------------
// Elementwise / reduction helper kernels
// ---------------------------------------------------------------------------
__global__ __launch_bounds__(256)
void layernorm_kernel(const float* __restrict__ x, const float* __restrict__ w,
                      const float* __restrict__ bsc, float* __restrict__ y)
{
    __shared__ float s1[256], s2[256];
    const int row = blockIdx.x, t = threadIdx.x;
    const float* xr = x + (size_t)row * Hc;
    float a = 0.0f, q = 0.0f;
    for (int j = t; j < Hc; j += 256) { float v = xr[j]; a += v; q += v * v; }
    s1[t] = a; s2[t] = q; __syncthreads();
    for (int off = 128; off > 0; off >>= 1) {
        if (t < off) { s1[t] += s1[t + off]; s2[t] += s2[t + off]; }
        __syncthreads();
    }
    float mean = s1[0] * (1.0f / Hc);
    float var  = s2[0] * (1.0f / Hc) - mean * mean;
    float inv  = rsqrtf(var + EPSc);
    for (int j = t; j < Hc; j += 256)
        y[(size_t)row * Hc + j] = (xr[j] - mean) * inv * w[j] + bsc[j];
}

__global__ __launch_bounds__(256)
void select_kernel(const float* __restrict__ arec, const float* __restrict__ abase,
                   const int* __restrict__ li, float* __restrict__ y, int nTotal)
{
    int i = blockIdx.x * 256 + threadIdx.x;
    if (i >= nTotal) return;
    int l = li[0];
    bool rec = (l % 4 == 0) && (l > 0);
    y[i] = rec ? arec[i] : abase[i];
}

__global__ __launch_bounds__(256)
void pool_kernel(const float* __restrict__ h, float* __restrict__ pooled)
{
    int idx = blockIdx.x * 256 + threadIdx.x;   // Bc*Hc threads total
    if (idx >= Bc * Hc) return;
    int b = idx / Hc, j = idx % Hc;
    const float* p = h + (size_t)b * Sc * Hc + j;
    float s = 0.0f;
    for (int si = 0; si < Sc; ++si) s += p[(size_t)si * Hc];
    pooled[idx] = s * (1.0f / Sc);
}

__global__ __launch_bounds__(128)
void matvec_kernel(const float* __restrict__ x, const float* __restrict__ W,
                   const float* __restrict__ bias, float* __restrict__ y,
                   int K, int N, int act)
{
    __shared__ float sh[128];
    const int n = blockIdx.x, b = blockIdx.y, t = threadIdx.x;
    const float* xr = x + (size_t)b * K;
    const float* wr = W + (size_t)n * K;
    float s = 0.0f;
    for (int k = t; k < K; k += 128) s += xr[k] * wr[k];
    sh[t] = s; __syncthreads();
    for (int off = 64; off > 0; off >>= 1) {
        if (t < off) sh[t] += sh[t + off];
        __syncthreads();
    }
    if (t == 0) {
        float v = sh[0] + (bias ? bias[n] : 0.0f);
        if (act == 3) v = 1.0f / (1.0f + __expf(-v));
        y[(size_t)b * N + n] = v;
    }
}

__global__ __launch_bounds__(128)
void softmax128_kernel(float* __restrict__ x)
{
    __shared__ float sh[128];
    const int b = blockIdx.x, t = threadIdx.x;
    float v = x[b * MBANKc + t];
    sh[t] = v; __syncthreads();
    for (int off = 64; off > 0; off >>= 1) {
        if (t < off) sh[t] = fmaxf(sh[t], sh[t + off]);
        __syncthreads();
    }
    float mx = sh[0]; __syncthreads();
    float e = __expf(v - mx);
    sh[t] = e; __syncthreads();
    for (int off = 64; off > 0; off >>= 1) {
        if (t < off) sh[t] += sh[t + off];
        __syncthreads();
    }
    x[b * MBANKc + t] = e / sh[0];
}

__global__ __launch_bounds__(256)
void retrieve_kernel(const float* __restrict__ rw, const float* __restrict__ mem,
                     float* __restrict__ outv)
{
    int i = blockIdx.x * 256 + threadIdx.x;   // Bc*Hc
    if (i >= Bc * Hc) return;
    int b = i / Hc, j = i % Hc;
    float s = 0.0f;
    for (int m = 0; m < MBANKc; ++m) s += rw[b * MBANKc + m] * mem[(size_t)m * Hc + j];
    outv[i] = s;
}

__global__ __launch_bounds__(256)
void addmem_kernel(float* __restrict__ h, const float* __restrict__ retrieved,
                   const float* __restrict__ scales, const int* __restrict__ li)
{
    size_t i = (size_t)blockIdx.x * 256 + threadIdx.x;   // Bc*Sc*Hc
    float sc = scales[li[0]];
    int j = (int)(i % Hc);
    int b = (int)(i / ((size_t)Sc * Hc));
    h[i] += retrieved[(size_t)b * Hc + j] * sc;
}

__global__ __launch_bounds__(256)
void pooled2_kernel(const float* __restrict__ pooled, const float* __restrict__ retrieved,
                    const float* __restrict__ scales, const int* __restrict__ li,
                    float* __restrict__ p2)
{
    int i = blockIdx.x * 256 + threadIdx.x;   // Bc*Hc
    if (i >= Bc * Hc) return;
    p2[i] = pooled[i] + retrieved[i] * scales[li[0]];
}

__global__ void topk_kernel(const float* __restrict__ aw, int* __restrict__ idx)
{
    int b = blockIdx.x;
    if (threadIdx.x != 0) return;
    float vals[MBANKc];
    for (int m = 0; m < MBANKc; ++m) vals[m] = aw[b * MBANKc + m];
    for (int t = 0; t < TOPKc; ++t) {
        int am = 0; float bv = vals[0];
        for (int m = 1; m < MBANKc; ++m)
            if (vals[m] > bv) { bv = vals[m]; am = m; }
        idx[b * TOPKc + t] = am;
        vals[am] = -3.4e38f;
    }
}

__global__ __launch_bounds__(256)
void scatter_kernel(const float* __restrict__ mem, const int* __restrict__ idx,
                    const float* __restrict__ upd, float* __restrict__ outm)
{
    const int m = blockIdx.x;
    __shared__ int win;
    if (threadIdx.x == 0) {
        int w = -1;
        for (int b = 0; b < Bc; ++b)
            for (int t = 0; t < TOPKc; ++t)
                if (idx[b * TOPKc + t] == m) w = b;   // later batch wins
        win = w;
    }
    __syncthreads();
    int w = win;
    for (int j = threadIdx.x; j < Hc; j += 256)
        outm[(size_t)m * Hc + j] = (w >= 0) ? upd[(size_t)w * Hc + j]
                                            : mem[(size_t)m * Hc + j];
}

// ---------------------------------------------------------------------------
extern "C" void kernel_launch(void* const* d_in, const int* in_sizes, int n_in,
                              void* d_out, int out_size, void* d_ws, size_t ws_size,
                              hipStream_t stream)
{
    (void)in_sizes; (void)n_in; (void)out_size; (void)ws_size;

    const float* hidden    = (const float*)d_in[0];
    const float* ln1_w     = (const float*)d_in[1];
    const float* ln1_b     = (const float*)d_in[2];
    const float* ln2_w     = (const float*)d_in[3];
    const float* ln2_b     = (const float*)d_in[4];
    const float* rec_wq    = (const float*)d_in[5];
    const float* rec_wk    = (const float*)d_in[6];
    const float* rec_wv    = (const float*)d_in[7];
    const float* rec_wo    = (const float*)d_in[8];
    const float* rec_ad_w1 = (const float*)d_in[9];
    const float* rec_ad_w2 = (const float*)d_in[10];
    const float* mem_bank  = (const float*)d_in[11];
    const float* alloc_w   = (const float*)d_in[12];
    const float* alloc_b   = (const float*)d_in[13];
    const float* retr_w    = (const float*)d_in[14];
    const float* retr_b    = (const float*)d_in[15];
    const float* upd_w     = (const float*)d_in[16];
    const float* upd_b     = (const float*)d_in[17];
    const float* mem_scales= (const float*)d_in[18];
    const float* attn_in_w = (const float*)d_in[19];
    const float* attn_in_b = (const float*)d_in[20];
    const float* attn_out_w= (const float*)d_in[21];
    const float* attn_out_b= (const float*)d_in[22];
    const float* mlp_w1    = (const float*)d_in[23];
    const float* mlp_b1    = (const float*)d_in[24];
    const float* mlp_w2    = (const float*)d_in[25];
    const float* mlp_b2    = (const float*)d_in[26];
    const int*   layer_idx = (const int*)d_in[27];

    float* out_main = (float*)d_out;                               // [B,S,H]
    float* out_mem  = out_main + (size_t)Bc * Sc * Hc;             // [MBANK,H]

    const size_t M = (size_t)Bc * Sc;     // 8192 rows
    float* ws = (float*)d_ws;
    size_t off = 0;
    float* hln   = ws + off; off += M * Hc;          // ln1 output / later h3
    float* qkvb  = ws + off; off += M * 3 * Hc;      // qkv scratch
    float* bufB  = ws + off; off += M * Hc;          // sdpa out / h_rec
    float* bufC  = ws + off; off += M * Hc;          // base_out / h / ln2
    float* mid   = ws + off; off += M * FFc;         // adapter mid + mlp mid
    float* pooled    = ws + off; off += Bc * Hc;
    float* alloc_l   = ws + off; off += Bc * MBANKc;
    float* retr_l    = ws + off; off += Bc * MBANKc;
    float* retrieved = ws + off; off += Bc * Hc;
    float* pooled2   = ws + off; off += Bc * Hc;
    float* upd_sig   = ws + off; off += Bc * Hc;
    int*   topidx    = (int*)(ws + off); off += Bc * TOPKc;

    const int H3 = 3 * Hc;
    dim3 gemmH(Hc / GT_N, (int)(M / GT_M));      // N=1024
    dim3 gemm3H(H3 / GT_N, (int)(M / GT_M));     // N=3072
    dim3 gemmAd(256 / GT_N, (int)(M / GT_M));    // N=256
    dim3 gemmFF(FFc / GT_N, (int)(M / GT_M));    // N=4096
    dim3 attnGrid(Sc / 16, NHc, Bc);

    // 1. ln1
    layernorm_kernel<<<(int)M, 256, 0, stream>>>(hidden, ln1_w, ln1_b, hln);

    // 2-4. recycler q/k/v projections into qkv buffer columns [0,H,2H)
    gemm_bf16_wmma<<<gemmH, 128, 0, stream>>>(hln, Hc, rec_wq, Hc, nullptr, nullptr, 0, qkvb + 0,      H3, Hc, 0);
    gemm_bf16_wmma<<<gemmH, 128, 0, stream>>>(hln, Hc, rec_wk, Hc, nullptr, nullptr, 0, qkvb + Hc,     H3, Hc, 0);
    gemm_bf16_wmma<<<gemmH, 128, 0, stream>>>(hln, Hc, rec_wv, Hc, nullptr, nullptr, 0, qkvb + 2 * Hc, H3, Hc, 0);

    // 5. recycler SDPA
    flash_attn_wmma<<<attnGrid, 32, 0, stream>>>(qkvb, H3, 0, Hc, 2 * Hc, bufB, Hc);

    // 6. base_out = sdpa @ rec_wo^T
    gemm_bf16_wmma<<<gemmH, 128, 0, stream>>>(bufB, Hc, rec_wo, Hc, nullptr, nullptr, 0, bufC, Hc, Hc, 0);
    // 7. mid = relu(base_out @ ad_w1^T)     [M,256]
    gemm_bf16_wmma<<<gemmAd, 128, 0, stream>>>(bufC, Hc, rec_ad_w1, Hc, nullptr, nullptr, 0, mid, 256, Hc, 1);
    // 8. h_rec = mid @ ad_w2^T              [M,H]
    gemm_bf16_wmma<<<gemmH, 128, 0, stream>>>(mid, 256, rec_ad_w2, 256, nullptr, nullptr, 0, bufB, Hc, 256, 0);

    // 9. h = recycle? h_rec : h_ln   -> bufC
    {
        int nTot = (int)(M * Hc);
        select_kernel<<<(nTot + 255) / 256, 256, 0, stream>>>(bufB, hln, layer_idx, bufC, nTot);
    }

    // 10. pooled = mean_s(h)
    pool_kernel<<<(Bc * Hc + 255) / 256, 256, 0, stream>>>(bufC, pooled);

    // 11-13. alloc / retr logits + softmax
    {
        dim3 ga(MBANKc, Bc);
        matvec_kernel<<<ga, 128, 0, stream>>>(pooled, alloc_w, alloc_b, alloc_l, Hc, MBANKc, 0);
        matvec_kernel<<<ga, 128, 0, stream>>>(pooled, retr_w,  retr_b,  retr_l,  Hc, MBANKc, 0);
        softmax128_kernel<<<Bc, 128, 0, stream>>>(alloc_l);
        softmax128_kernel<<<Bc, 128, 0, stream>>>(retr_l);
    }

    // 14. retrieved = retr_soft @ mem_bank
    retrieve_kernel<<<(Bc * Hc + 255) / 256, 256, 0, stream>>>(retr_l, mem_bank, retrieved);

    // 15. h += retrieved * scale ; pooled2 = pooled + retrieved * scale
    addmem_kernel<<<(int)(M * Hc / 256), 256, 0, stream>>>(bufC, retrieved, mem_scales, layer_idx);
    pooled2_kernel<<<(Bc * Hc + 255) / 256, 256, 0, stream>>>(pooled, retrieved, mem_scales, layer_idx, pooled2);

    // 16. update_signals = sigmoid(pooled2 @ upd_w^T + upd_b)
    {
        dim3 gu(Hc, Bc);
        matvec_kernel<<<gu, 128, 0, stream>>>(pooled2, upd_w, upd_b, upd_sig, Hc, Hc, 3);
    }

    // 17-18. top-k + scatter into new_mem_bank (output region 2)
    topk_kernel<<<Bc, 32, 0, stream>>>(alloc_l, topidx);
    scatter_kernel<<<MBANKc, 256, 0, stream>>>(mem_bank, topidx, upd_sig, out_mem);

    // 19. self-attention qkv = h @ attn_in_w^T + attn_in_b   [M,3H]
    gemm_bf16_wmma<<<gemm3H, 128, 0, stream>>>(bufC, Hc, attn_in_w, Hc, attn_in_b, nullptr, 0, qkvb, H3, Hc, 0);

    // 20. SDPA
    flash_attn_wmma<<<attnGrid, 32, 0, stream>>>(qkvb, H3, 0, Hc, 2 * Hc, bufB, Hc);

    // 21. h3 = residual(hidden) + sdpa @ attn_out_w^T + attn_out_b   -> hln
    gemm_bf16_wmma<<<gemmH, 128, 0, stream>>>(bufB, Hc, attn_out_w, Hc, attn_out_b, hidden, Hc, hln, Hc, Hc, 0);

    // 22. ln2 -> bufC
    layernorm_kernel<<<(int)M, 256, 0, stream>>>(hln, ln2_w, ln2_b, bufC);

    // 23. mid = gelu(ln2 @ mlp_w1^T + b1)   [M,FF]
    gemm_bf16_wmma<<<gemmFF, 128, 0, stream>>>(bufC, Hc, mlp_w1, Hc, mlp_b1, nullptr, 0, mid, FFc, Hc, 2);

    // 24. out = h3 + mid @ mlp_w2^T + b2
    gemm_bf16_wmma<<<gemmH, 128, 0, stream>>>(mid, FFc, mlp_w2, FFc, mlp_b2, hln, Hc, out_main, Hc, FFc, 0);
}
